// MoE_60421599920489
// MI455X (gfx1250) — compile-verified, matrix operand order checked by reading, not run on hardware
//
#include <hip/hip_runtime.h>
#include <hip/hip_bf16.h>

// ---------------------------------------------------------------------------
// MoE (router MLP + top-2 dispatch + expert FFN + gated combine) for MI455X.
//  - All GEMMs: v_wmma_f32_16x16x32_bf16, wave32, 128x128 block tiles.
//  - Weights pre-converted f32->bf16 and pre-transposed to [N][K] once
//    (halves GEMM HBM traffic; tile staging becomes contiguous 16B copies).
//  - Tile staging uses GLOBAL_LOAD_ASYNC_TO_LDS_B128 (ASYNCcnt) with double
//    buffering when the toolchain exposes the builtins; sync fallback else.
// Workspace (~332MB) with region reuse; all launches on `stream`.
// ---------------------------------------------------------------------------

#define BB 8
#define TT 2048
#define CC 1024
#define HH 4096
#define EE 8
#define KK 2
#define CAPP 320          // int(T/E*1.25)

typedef __attribute__((ext_vector_type(16))) __bf16 v16bf;
typedef __attribute__((ext_vector_type(8)))  __bf16 v8bf;
typedef __attribute__((ext_vector_type(4)))  __bf16 v4bf;
typedef __attribute__((ext_vector_type(8)))  float  v8f;
typedef __attribute__((ext_vector_type(4)))  float  v4f;
typedef __attribute__((ext_vector_type(4)))  int    v4i;

union FragAB { v16bf v; v8bf h[2]; };

__device__ __forceinline__ __bf16 f2bf(float f) { return (__bf16)f; }
__device__ __forceinline__ float  bf2f(__bf16 h) { return (float)h; }

// ---- CDNA5 async memory->LDS path (guarded; sync fallback otherwise) ------
#ifdef __has_builtin
#if __has_builtin(__builtin_amdgcn_global_load_async_to_lds_b128)
#if __has_builtin(__builtin_amdgcn_s_wait_asynccnt)
#define ASYNC_OK 1
#endif
#endif
#endif
#ifndef ASYNC_OK
#define ASYNC_OK 0
#endif

typedef __attribute__((address_space(1))) v4i* gptr_b128;
typedef __attribute__((address_space(3))) v4i* lptr_b128;

__device__ __forceinline__ void cp16_to_lds(const __bf16* g, __bf16* l)
{
#if ASYNC_OK
    __builtin_amdgcn_global_load_async_to_lds_b128(
        (gptr_b128)g, (lptr_b128)l, /*offset=*/0, /*cpol=*/0);
#else
    *(v8bf*)l = *(const v8bf*)g;
#endif
}

__device__ __forceinline__ void wait_stage()
{
#if ASYNC_OK
    __builtin_amdgcn_s_wait_asynccnt(0);
#endif
}

// ---------------------------------------------------------------------------
// Tiled WMMA GEMM:  C[M,N] = act( A[M,K] * Bt[N,K]^T + bias[N] )
// A, Bt are bf16; Bt is pre-transposed so both tiles stage identically.
// Block = 256 threads (8 wave32 waves); tile 128x128, K-tile 32, double-
// buffered LDS (40KB). Waves 4(M) x 2(N), each wave 32x64 = 2x4 WMMA accs.
// ---------------------------------------------------------------------------
template <typename OT, bool RELU>
__global__ __launch_bounds__(256)
void gemm_wmma_k(const __bf16* __restrict__ A, const __bf16* __restrict__ Bt,
                 const float* __restrict__ bias, OT* __restrict__ Cc,
                 int M, int N, int Kd)
{
    __shared__ __align__(16) __bf16 lA[2][128][40];   // [buf][m][k], 80B rows
    __shared__ __align__(16) __bf16 lB[2][128][40];   // [buf][n][k]

    const int tid  = threadIdx.x;
    const int bn   = blockIdx.x * 128;
    const int bm   = blockIdx.y * 128;
    const int wave = tid >> 5;
    const int lane = tid & 31;
    const int wm   = (wave & 3) * 32;   // wave M offset
    const int wn   = (wave >> 2) * 64;  // wave N offset
    const int lrow = lane & 15;
    const int sel  = lane >> 4;         // 0: K{0..7,16..23}, 1: K{8..15,24..31}

    v8f acc[2][4] = {};

    // Stage one 128x32 A tile + 128x32 B tile (16B chunks; 4 per thread).
    auto stage = [&](int bufi, int k0) {
        #pragma unroll
        for (int it = 0; it < 2; ++it) {
            int ch = tid + it * 256;        // 512 chunks per tile
            int r  = ch >> 2;               // 4 chunks per row
            int c8 = (ch & 3) * 8;
            cp16_to_lds(A  + (size_t)(bm + r) * Kd + k0 + c8, &lA[bufi][r][c8]);
            cp16_to_lds(Bt + (size_t)(bn + r) * Kd + k0 + c8, &lB[bufi][r][c8]);
        }
    };

    const int KT = Kd >> 5;
    int cur = 0;
    stage(0, 0);

    for (int kt = 0; kt < KT; ++kt) {
        wait_stage();          // tile kt arrived (this wave's async copies)
        __syncthreads();       // everyone's copies done; prior reads done
        if (kt + 1 < KT) stage(cur ^ 1, (kt + 1) << 5);

        FragAB a[2], b[4];
        #pragma unroll
        for (int mi = 0; mi < 2; ++mi) {
            int r = wm + mi * 16 + lrow;
            a[mi].h[0] = *(const v8bf*)&lA[cur][r][sel * 8];
            a[mi].h[1] = *(const v8bf*)&lA[cur][r][16 + sel * 8];
        }
        #pragma unroll
        for (int ni = 0; ni < 4; ++ni) {
            int c = wn + ni * 16 + lrow;
            b[ni].h[0] = *(const v8bf*)&lB[cur][c][sel * 8];
            b[ni].h[1] = *(const v8bf*)&lB[cur][c][16 + sel * 8];
        }
        #pragma unroll
        for (int mi = 0; mi < 2; ++mi)
            #pragma unroll
            for (int ni = 0; ni < 4; ++ni)
                acc[mi][ni] = __builtin_amdgcn_wmma_f32_16x16x32_bf16(
                    false, a[mi].v, false, b[ni].v,
                    (short)0, acc[mi][ni], false, false);
        cur ^= 1;
    }

    // Epilogue: D layout VGPR i -> M = i + 8*(lane>=16), N = lane%16.
    #pragma unroll
    for (int mi = 0; mi < 2; ++mi) {
        #pragma unroll
        for (int ni = 0; ni < 4; ++ni) {
            int col = bn + wn + ni * 16 + (lane & 15);
            float bv = bias ? bias[col] : 0.0f;
            #pragma unroll
            for (int i = 0; i < 8; ++i) {
                int row = bm + wm + mi * 16 + i + (lane >> 4) * 8;
                float v = acc[mi][ni][i] + bv;
                if (RELU) v = v > 0.0f ? v : 0.0f;
                Cc[(size_t)row * N + col] = (OT)v;
            }
        }
    }
}

// ---------------------------------------------------------------------------
// Convert f32 -> bf16 (contiguous), 4 elements/thread.
// ---------------------------------------------------------------------------
__global__ __launch_bounds__(256)
void cvt_bf16_k(const float* __restrict__ src, __bf16* __restrict__ dst, size_t n)
{
    size_t i = ((size_t)blockIdx.x * 256 + threadIdx.x) * 4;
    if (i >= n) return;
    v4f v = *(const v4f*)(src + i);
    v4bf o = { f2bf(v.x), f2bf(v.y), f2bf(v.z), f2bf(v.w) };
    *(v4bf*)(dst + i) = o;
}

// ---------------------------------------------------------------------------
// Transpose + convert: W[R][Cn] f32 (batched via blockIdx.z) -> Wt[Cn][R] bf16.
// 64x64 LDS tiles; both global accesses coalesced.
// ---------------------------------------------------------------------------
__global__ __launch_bounds__(256)
void transpose_cvt_k(const float* __restrict__ W, __bf16* __restrict__ Wt,
                     int R, int Cn, size_t inStride, size_t outStride)
{
    __shared__ __bf16 t[64][72];
    const float* Wb  = W  + (size_t)blockIdx.z * inStride;
    __bf16*      Wtb = Wt + (size_t)blockIdx.z * outStride;
    int r0 = blockIdx.y * 64, c0 = blockIdx.x * 64;

    #pragma unroll
    for (int it = 0; it < 4; ++it) {
        int idx = threadIdx.x + it * 256;   // 1024 chunks of 4 floats
        int r = idx >> 4;
        int c = (idx & 15) * 4;
        v4f v = *(const v4f*)(Wb + (size_t)(r0 + r) * Cn + c0 + c);
        t[c + 0][r] = f2bf(v.x);
        t[c + 1][r] = f2bf(v.y);
        t[c + 2][r] = f2bf(v.z);
        t[c + 3][r] = f2bf(v.w);
    }
    __syncthreads();
    #pragma unroll
    for (int it = 0; it < 4; ++it) {
        int idx = threadIdx.x + it * 256;
        int c = idx >> 4;
        int r = (idx & 15) * 4;
        v4bf o = { t[c][r], t[c][r + 1], t[c][r + 2], t[c][r + 3] };
        *(v4bf*)(Wtb + (size_t)(c0 + c) * R + r0 + r) = o;
    }
}

// ---------------------------------------------------------------------------
// Router head: logits = H2 @ W3 + b3, softmax over E=8, top-2 (lower-index
// tie-break, matching jax.lax.top_k). One wave per token.
// ---------------------------------------------------------------------------
__global__ __launch_bounds__(256)
void router_topk_k(const __bf16* __restrict__ H2, const float* __restrict__ W3,
                   const float* __restrict__ b3, int* __restrict__ topk_e,
                   float* __restrict__ gates, int Ntok, int Kd)
{
    int wave = threadIdx.x >> 5, lane = threadIdx.x & 31;
    int t = blockIdx.x * 8 + wave;
    if (t >= Ntok) return;

    float acc[8] = {};
    const __bf16* h = H2 + (size_t)t * Kd;
    for (int k = lane; k < Kd; k += 32) {
        float hv = bf2f(h[k]);
        const float* w = W3 + (size_t)k * EE;
        v4f w0 = *(const v4f*)w;
        v4f w1 = *(const v4f*)(w + 4);
        acc[0] += hv * w0.x; acc[1] += hv * w0.y;
        acc[2] += hv * w0.z; acc[3] += hv * w0.w;
        acc[4] += hv * w1.x; acc[5] += hv * w1.y;
        acc[6] += hv * w1.z; acc[7] += hv * w1.w;
    }
    #pragma unroll
    for (int e = 0; e < EE; ++e)
        #pragma unroll
        for (int off = 16; off > 0; off >>= 1)
            acc[e] += __shfl_xor(acc[e], off, 32);

    if (lane == 0) {
        float lg[EE], mx = -3.4e38f;
        #pragma unroll
        for (int e = 0; e < EE; ++e) { lg[e] = acc[e] + b3[e]; mx = lg[e] > mx ? lg[e] : mx; }
        float s = 0.0f, p[EE];
        #pragma unroll
        for (int e = 0; e < EE; ++e) { p[e] = __expf(lg[e] - mx); s += p[e]; }
        float inv = 1.0f / s;
        int e0 = 0, e1 = 0; float p0 = -1.0f, p1 = -1.0f;
        #pragma unroll
        for (int e = 0; e < EE; ++e) {
            float pe = p[e] * inv;
            if (pe > p0)      { e1 = e0; p1 = p0; e0 = e; p0 = pe; }
            else if (pe > p1) { e1 = e;  p1 = pe; }
        }
        topk_e[t * 2 + 0] = e0; topk_e[t * 2 + 1] = e1;
        gates [t * 2 + 0] = p0; gates [t * 2 + 1] = p1;
    }
}

// ---------------------------------------------------------------------------
// Deterministic capacity scan: one thread per (b,e); sequential cumsum over T
// (mirrors the reference cumsum + capacity truncation exactly).
// ---------------------------------------------------------------------------
__global__ void route_scan_k(const int* __restrict__ topk_e,
                             int* __restrict__ slot_of, int* __restrict__ token_of)
{
    int id = threadIdx.x;
    if (id >= BB * EE) return;
    int b = id >> 3, e = id & 7;
    int cnt = 0;
    for (int t = 0; t < TT; ++t) {
        int gi = (b * TT + t) * 2;
        #pragma unroll
        for (int k = 0; k < KK; ++k) {
            if (topk_e[gi + k] == e) {
                if (cnt < CAPP) {
                    slot_of[gi + k] = cnt;
                    token_of[((size_t)e * BB + b) * CAPP + cnt] = t;
                } else {
                    slot_of[gi + k] = -1;
                }
                cnt++;
            }
        }
    }
    int filled = cnt < CAPP ? cnt : CAPP;
    for (int s = filled; s < CAPP; ++s)
        token_of[((size_t)e * BB + b) * CAPP + s] = -1;
}

// ---------------------------------------------------------------------------
// Dispatch: gather x rows into [E][B][CAP][C] bf16 buffer (zeros if empty).
// ---------------------------------------------------------------------------
__global__ __launch_bounds__(256)
void dispatch_k(const float* __restrict__ x, const int* __restrict__ token_of,
                __bf16* __restrict__ buf)
{
    int row = blockIdx.x;               // (e*B + b)*CAP + s
    int b = (row / CAPP) & (BB - 1);
    int t = token_of[row];
    __bf16* dst = buf + (size_t)row * CC;
    int c = threadIdx.x * 4;
    if (t < 0) {
        v4bf z = { f2bf(0.f), f2bf(0.f), f2bf(0.f), f2bf(0.f) };
        *(v4bf*)(dst + c) = z;
    } else {
        const float* src = x + ((size_t)b * TT + t) * CC + c;
        v4f v = *(const v4f*)src;
        v4bf o = { f2bf(v.x), f2bf(v.y), f2bf(v.z), f2bf(v.w) };
        *(v4bf*)(dst + c) = o;
    }
}

// ---------------------------------------------------------------------------
// Combine: out[b,t,:] = sum_k gate_k * y[e_k][b][slot_k][:].
// ---------------------------------------------------------------------------
__global__ __launch_bounds__(256)
void combine_k(const float* __restrict__ ybuf, const int* __restrict__ topk_e,
               const int* __restrict__ slot_of, const float* __restrict__ gates,
               float* __restrict__ out)
{
    int bt = blockIdx.x;                // b*T + t
    int b  = bt / TT;
    int c  = threadIdx.x * 4;
    float a0 = 0.f, a1 = 0.f, a2 = 0.f, a3 = 0.f;
    #pragma unroll
    for (int k = 0; k < KK; ++k) {
        int e = topk_e[bt * 2 + k];
        int s = slot_of[bt * 2 + k];
        float g = gates[bt * 2 + k];
        if (s >= 0) {
            const float* y = ybuf + (((size_t)e * BB + b) * CAPP + s) * CC + c;
            v4f v = *(const v4f*)y;
            a0 += g * v.x; a1 += g * v.y; a2 += g * v.z; a3 += g * v.w;
        }
    }
    float* o = out + (size_t)bt * CC + c;
    o[0] = a0; o[1] = a1; o[2] = a2; o[3] = a3;
}

// ---------------------------------------------------------------------------
// Host pipeline. Workspace layout with time-based reuse (~332MB):
//   [  0, 32)MB  xbf (bf16 x)          -> after gemm1: hbuf (20MB)
//   [ 32, 40)MB  wT1
//   [ 40, 72)MB  wT2
//   [ 72,200)MB  h1                    -> after gemm2: ewT1 (64MB) + ewT2 (64MB)
//   [200,328)MB  h2                    -> after router: ybuf (80MB) + buf (40MB)
//   [328,332)MB  routing metadata
// ---------------------------------------------------------------------------
extern "C" void kernel_launch(void* const* d_in, const int* in_sizes, int n_in,
                              void* d_out, int out_size, void* d_ws, size_t ws_size,
                              hipStream_t stream)
{
    const float* x   = (const float*)d_in[0];
    const float* rw1 = (const float*)d_in[1];
    const float* rb1 = (const float*)d_in[2];
    const float* rw2 = (const float*)d_in[3];
    const float* rb2 = (const float*)d_in[4];
    const float* rw3 = (const float*)d_in[5];
    const float* rb3 = (const float*)d_in[6];
    const float* ew1 = (const float*)d_in[7];
    const float* eb1 = (const float*)d_in[8];
    const float* ew2 = (const float*)d_in[9];
    const float* eb2 = (const float*)d_in[10];
    float* out = (float*)d_out;

    char* ws = (char*)d_ws;
    const size_t MB = 1024ull * 1024ull;
    __bf16* xbf   = (__bf16*)(ws + 0);
    __bf16* hbuf  = (__bf16*)(ws + 0);          // reuse xbf region
    __bf16* wT1   = (__bf16*)(ws + 32 * MB);
    __bf16* wT2   = (__bf16*)(ws + 40 * MB);
    __bf16* h1    = (__bf16*)(ws + 72 * MB);
    __bf16* ewT1  = (__bf16*)(ws + 72 * MB);    // reuse h1 region
    __bf16* ewT2  = (__bf16*)(ws + 136 * MB);   // reuse h1 region
    __bf16* h2    = (__bf16*)(ws + 200 * MB);
    float*  ybuf  = (float*) (ws + 200 * MB);   // reuse h2 region
    __bf16* buf   = (__bf16*)(ws + 280 * MB);   // reuse h2 region
    int*    topk  = (int*)   (ws + 328 * MB);
    float*  gates = (float*) (ws + 329 * MB);
    int*    slot  = (int*)   (ws + 330 * MB);
    int*    tokof = (int*)   (ws + 331 * MB);

    const int Ntok = BB * TT;        // 16384
    const int Mexp = BB * CAPP;      // 2560

    // --- one-time conversions: x -> bf16; router weights -> bf16 [N][K] ---
    cvt_bf16_k<<<(Ntok * CC) / (256 * 4), 256, 0, stream>>>(x, xbf, (size_t)Ntok * CC);
    transpose_cvt_k<<<dim3(HH / 64, CC / 64, 1), 256, 0, stream>>>(
        rw1, wT1, CC, HH, 0, 0);
    transpose_cvt_k<<<dim3(HH / 64, HH / 64, 1), 256, 0, stream>>>(
        rw2, wT2, HH, HH, 0, 0);

    // --- router MLP (bf16 WMMA, f32 accumulate) ---
    gemm_wmma_k<__bf16, true>
        <<<dim3(HH / 128, Ntok / 128), 256, 0, stream>>>(xbf, wT1, rb1, h1, Ntok, HH, CC);
    gemm_wmma_k<__bf16, true>
        <<<dim3(HH / 128, Ntok / 128), 256, 0, stream>>>(h1, wT2, rb2, h2, Ntok, HH, HH);

    // --- router head: logits -> softmax -> top-2 ---
    router_topk_k<<<Ntok / 8, 256, 0, stream>>>(h2, rw3, rb3, topk, gates, Ntok, HH);

    // --- expert weights -> bf16 [N][K] (reuses h1 region; h1 now dead) ---
    transpose_cvt_k<<<dim3(HH / 64, CC / 64, EE), 256, 0, stream>>>(
        ew1, ewT1, CC, HH, (size_t)CC * HH, (size_t)CC * HH);
    transpose_cvt_k<<<dim3(CC / 64, HH / 64, EE), 256, 0, stream>>>(
        ew2, ewT2, HH, CC, (size_t)HH * CC, (size_t)HH * CC);

    // --- routing: capacity scan + dispatch ---
    route_scan_k<<<1, 64, 0, stream>>>(topk, slot, tokof);
    dispatch_k<<<EE * BB * CAPP, 256, 0, stream>>>(x, tokof, buf);

    // --- expert FFNs (hbuf region reused per expert) ---
    for (int e = 0; e < EE; ++e) {
        const __bf16* be = buf + (size_t)e * Mexp * CC;
        gemm_wmma_k<__bf16, true>
            <<<dim3(HH / 128, Mexp / 128), 256, 0, stream>>>(
                be, ewT1 + (size_t)e * CC * HH, eb1 + (size_t)e * HH, hbuf,
                Mexp, HH, CC);
        gemm_wmma_k<float, false>
            <<<dim3(CC / 128, Mexp / 128), 256, 0, stream>>>(
                hbuf, ewT2 + (size_t)e * HH * CC, eb2 + (size_t)e * CC,
                ybuf + (size_t)e * Mexp * CC, Mexp, CC, HH);
    }

    // --- gated combine back to token order ---
    combine_k<<<Ntok, 256, 0, stream>>>(ybuf, topk, slot, gates, out);
}
